// DeformableConv2d_65781719105717
// MI455X (gfx1250) — matrix operand order repaired and checked by reading, not compiled, not run on hardware
//
#include <hip/hip_runtime.h>

// ---------------------------------------------------------------------------
// Types for gfx1250 WMMA (wave32): v_wmma_f32_16x16x32_bf16
// ---------------------------------------------------------------------------
typedef __bf16 bf16;
typedef bf16  v8bf  __attribute__((ext_vector_type(8)));
typedef bf16  v16bf __attribute__((ext_vector_type(16)));
typedef float v8f   __attribute__((ext_vector_type(8)));

#define BATCH   8
#define CHW     4096        // 64*64
#define NTOT    32768       // BATCH * CHW
#define KTOT    2304        // CIN*9, ordered kg = 32*(chunk*9 + r) + (ci&31)
#define TILE_M  128
#define TILE_N  128
#define LDA     40          // lA row stride in bf16 elems (80B, 16B aligned)
#define ABUFB   (TILE_M * LDA * 2)   // bytes per A buffer (10,240)
#define RAWROWS 4           // h0-1 .. h0+2
#define RAWW    66          // -1 .. 64 halo columns
#define RAWSTR  32          // ci entries per (row,col) cell

// ---------------------------------------------------------------------------
// x [B][CIN][64][64] fp32  ->  xb [CIN][B*4096] bf16  (channel-major planes)
// ---------------------------------------------------------------------------
__global__ __launch_bounds__(256) void k_convert_x(const float* __restrict__ x,
                                                   bf16* __restrict__ xb) {
    size_t i = (size_t)blockIdx.x * 256 + threadIdx.x;   // 8,388,608 total
    int hw = (int)(i & 4095);
    int ci = (int)((i >> 12) & 255);
    int b  = (int)(i >> 20);
    xb[(size_t)ci * NTOT + b * CHW + hw] = (bf16)x[i];
}

// ---------------------------------------------------------------------------
// w [CO][256][3][3] fp32 -> wb [CO][2304] bf16, kg = (ci>>5)*288 + r*32 + (ci&31)
// ---------------------------------------------------------------------------
__global__ __launch_bounds__(256) void k_convert_w(const float* __restrict__ w,
                                                   bf16* __restrict__ wb) {
    size_t i = (size_t)blockIdx.x * 256 + threadIdx.x;   // CO*2304 total
    int r  = (int)(i % 9);
    size_t q = i / 9;
    int ci = (int)(q & 255);
    int co = (int)(q >> 8);
    wb[(size_t)co * KTOT + (ci >> 5) * 288 + r * 32 + (ci & 31)] = (bf16)w[i];
}

// ---------------------------------------------------------------------------
// Implicit-GEMM 3x3 conv + fused BatchNorm, bf16 WMMA, fp32 accumulate.
// Block: 256 thr = 8 waves (2 M x 4 N), tile 128x128 (co x spatial).
// Flattened loop over 72 (chunk,tap) steps; x slab staged once per chunk and
// shared by 9 taps; A tile double-buffered via global_load_async_to_lds_b128
// with s_wait_asynccnt 2 so the DMA overlaps the WMMAs.
// Grid: (NTOT/128, CO/128)
// ---------------------------------------------------------------------------
__global__ __launch_bounds__(256) void k_conv_bn(
    const bf16* __restrict__ xb,     // [256][32768]
    const bf16* __restrict__ wb,     // [CO][2304]
    const float* __restrict__ gamma, const float* __restrict__ beta,
    const float* __restrict__ mean,  const float* __restrict__ var,
    float* __restrict__ out)         // [CO][32768] channel-major
{
    __shared__ __align__(16) bf16 raw[RAWROWS * RAWW * RAWSTR];  // 16,896 B
    __shared__ __align__(16) bf16 lA[2][TILE_M * LDA];           // 2 x 10,240 B

    const int t    = threadIdx.x;
    const int lane = t & 31;
    const int wv   = t >> 5;
    const int wm   = wv & 1;          // wave M slot (0..1) -> 64 rows
    const int wn   = wv >> 1;         // wave N slot (0..3) -> 32 cols

    const int s0  = blockIdx.x * TILE_N;   // spatial tile: rows h0,h0+1 of b
    const int co0 = blockIdx.y * TILE_M;
    const int b   = s0 >> 12;
    const int h0  = (s0 & 4095) >> 6;

    // raw-slab staging mapping: ci = t&31, (row, 32-col half) = t>>5
    const int sci   = t & 31;
    const int srow  = (t >> 5) >> 1;        // 0..3
    const int swseg = ((t >> 5) & 1) << 5;  // 0 or 32
    // w-halo zeroing mapping (256 elems: 4 rows x 2 cols x 32 ci)
    const int zrow = t >> 6;
    const int zcol = ((t >> 5) & 1) * 65;
    const int zci  = t & 31;

    // A staging mapping: 2 threads per row, 16 k each (32 B via 2 async b128)
    const int arow  = t >> 1;
    const int ahalf = (t & 1) << 4;
    const bf16* wrow = wb + (size_t)(co0 + arow) * KTOT + ahalf;
    const unsigned ldsA0 = (unsigned)(uintptr_t)&lA[0][arow * LDA + ahalf];

    // issue one tap's A tile (128co x 32k) into buffer `buf` asynchronously
    auto issueA = [&](int i, int buf) {
        const bf16* src = wrow + i * 32;
        unsigned dst = ldsA0 + (unsigned)buf * ABUFB;
        asm volatile(
            "global_load_async_to_lds_b128 %0, %1, off\n\t"
            "global_load_async_to_lds_b128 %0, %1, off offset:16"
            :: "v"(dst), "v"(src)
            : "memory");
    };

    v8f acc[4][2] = {};

    issueA(0, 0);
    int X = 0;
    for (int i = 0; i < 72; ++i) {
        __syncthreads();   // barrier1: prior readers of raw & spare buf done

        if ((i % 9) == 0) {
            // ---- stage raw x slab (halo'd, ci-innermost for b128 reads) ----
            const int ci = (i / 9) * 32 + sci;
            const bf16* plane = xb + (size_t)ci * NTOT + b * CHW;
            const int hsrc = h0 - 1 + srow;
            v8bf vals[4];
            if ((unsigned)hsrc < 64u) {
                const bf16* src = plane + hsrc * 64 + swseg;
                #pragma unroll
                for (int q = 0; q < 4; ++q) vals[q] = *(const v8bf*)(src + q * 8);
            } else {
                #pragma unroll
                for (int q = 0; q < 4; ++q)
                    #pragma unroll
                    for (int e = 0; e < 8; ++e) vals[q][e] = (bf16)0.0f;
            }
            #pragma unroll
            for (int e = 0; e < 32; ++e)
                raw[(srow * RAWW + 1 + swseg + e) * RAWSTR + sci] = vals[e >> 3][e & 7];
            raw[(zrow * RAWW + zcol) * RAWSTR + zci] = (bf16)0.0f;   // w halo
        }

        // prefetch next tap's A into the spare buffer; drain only current buf
        if (i + 1 < 72) {
            issueA(i + 1, X ^ 1);
            asm volatile("s_wait_asynccnt 0x2" ::: "memory");
        } else {
            asm volatile("s_wait_asynccnt 0x0" ::: "memory");
        }
        __syncthreads();   // barrier2: current A writes + raw stores visible

        // ---- fragments per ISA 16-bit layouts ----
        const int r = i % 9;
        const int rsel   = lane & 15;
        const int khalfA = (lane < 16) ? 0 : 8;   // A: lanes 16-31 hold K+8
        const int kstart = (lane < 16) ? 0 : 8;   // B: lanes 16-31 hold K+8
        const int dh3 = r / 3;                    // slab row offset (dh+1)
        const int dw3 = r % 3;                    // slab col offset (dw+1)
        const bf16* lAx = lA[X];

        v16bf afrag[4];
        #pragma unroll
        for (int ii = 0; ii < 4; ++ii) {
            int row = wm * 64 + ii * 16 + rsel;
            v8bf lo = *(const v8bf*)&lAx[row * LDA + khalfA];
            v8bf hi = *(const v8bf*)&lAx[row * LDA + 16 + khalfA];
            afrag[ii] = __builtin_shufflevector(lo, hi, 0,1,2,3,4,5,6,7,
                                                         8,9,10,11,12,13,14,15);
        }
        v16bf bfrag[2];
        #pragma unroll
        for (int j = 0; j < 2; ++j) {
            int col = wn * 32 + j * 16 + rsel;
            int ch  = col >> 6;
            int cw  = col & 63;
            const bf16* cell =
                &raw[((ch + dh3) * RAWW + cw + dw3) * RAWSTR + kstart];
            v8bf lo = *(const v8bf*)cell;
            v8bf hi = *(const v8bf*)(cell + 16);
            bfrag[j] = __builtin_shufflevector(lo, hi, 0,1,2,3,4,5,6,7,
                                                        8,9,10,11,12,13,14,15);
        }
        #pragma unroll
        for (int ii = 0; ii < 4; ++ii)
            #pragma unroll
            for (int j = 0; j < 2; ++j)
                acc[ii][j] = __builtin_amdgcn_wmma_f32_16x16x32_bf16(
                    false, afrag[ii], false, bfrag[j],
                    (short)0, acc[ii][j], false, false);
        X ^= 1;
    }

    // ---- epilogue: fused BN, store fp32 channel-major ----
    const int mofs = (lane < 16) ? 0 : 8;   // C/D: lanes 16-31 hold M+8
    const int ncol = lane & 15;
    #pragma unroll
    for (int ii = 0; ii < 4; ++ii) {
        #pragma unroll
        for (int v = 0; v < 8; ++v) {
            int co = co0 + wm * 64 + ii * 16 + v + mofs;
            float inv = rsqrtf(var[co] + 1e-5f);
            float sc  = gamma[co] * inv;
            float sh  = beta[co] - mean[co] * sc;
            #pragma unroll
            for (int j = 0; j < 2; ++j) {
                int s = s0 + wn * 32 + j * 16 + ncol;
                out[(size_t)co * NTOT + s] = acc[ii][j][v] * sc + sh;
            }
        }
    }
}

// ---------------------------------------------------------------------------
// Chained deformable gather + bilinear + ReLU.  One block per (b,c) plane.
// out2 reshape [B,512,H,W] -> [B,256,H,W,2] mixes channel & space:
//   off(b,c,h,w,e) = out2[2c + (h>>5)][b][ 128*(h&31) + 2w + e ]
// ---------------------------------------------------------------------------
__device__ __forceinline__ void calc_idx(const float* ob, int p,
                                         int& i1, int& i2, int& i3, int& i4,
                                         float& d1, float& d2, float& d3, float& d4) {
    int h = p >> 6, w = p & 63;
    int q = ((h >> 5) << 15) + ((h & 31) << 7) + (w << 1);
    float ox = ob[q], oy = ob[q + 1];
    float gx = fminf(fmaxf((float)w + ox, 0.0f), 63.0f);
    float gy = fminf(fmaxf((float)h + oy, 0.0f), 63.0f);
    float fx = floorf(gx), cx = ceilf(gx);
    float fy = floorf(gy), cy = ceilf(gy);
    d1 = gx - fx; d2 = cx - gx; d3 = gy - fy; d4 = cy - gy;
    int fxi = (int)fx, cxi = (int)cx, fyi = (int)fy, cyi = (int)cy;
    i1 = fxi * 64 + fyi;    // reference uses ix*W + iy (column-major quirk)
    i2 = fxi * 64 + cyi;
    i3 = cxi * 64 + cyi;
    i4 = cxi * 64 + fyi;
}

__global__ __launch_bounds__(256) void k_deform(const float* __restrict__ out1,
                                                const float* __restrict__ out2,
                                                float* __restrict__ dst) {
    __shared__ float fm[CHW];
    __shared__ float va[CHW];
    __shared__ float vb[CHW];
    __shared__ float vc[CHW];

    const int bc = blockIdx.x;      // 0..2047
    const int b  = bc >> 8;
    const int c  = bc & 255;
    const float* f1 = out1 + (size_t)c * NTOT + b * CHW;
    const float* ob = out2 + (size_t)(2 * c) * NTOT + b * CHW;
    const int t = threadIdx.x;

    int i1, i2, i3, i4; float d1, d2, d3, d4;

    #pragma unroll
    for (int e = 0; e < 16; ++e) fm[e * 256 + t] = f1[e * 256 + t];
    __syncthreads();

    #pragma unroll 4
    for (int e = 0; e < 16; ++e) {
        int p = e * 256 + t;
        calc_idx(ob, p, i1, i2, i3, i4, d1, d2, d3, d4);
        va[p] = fm[i1];                         // v1 = gather(out1, fx, fy)
    }
    __syncthreads();
    #pragma unroll 4
    for (int e = 0; e < 16; ++e) {
        int p = e * 256 + t;
        calc_idx(ob, p, i1, i2, i3, i4, d1, d2, d3, d4);
        vb[p] = va[i2];                         // v2 = gather(v1, fx, cy)
    }
    __syncthreads();
    #pragma unroll 4
    for (int e = 0; e < 16; ++e) {
        int p = e * 256 + t;
        calc_idx(ob, p, i1, i2, i3, i4, d1, d2, d3, d4);
        vc[p] = vb[i3];                         // v3 = gather(v2, cx, cy)
    }
    __syncthreads();
    #pragma unroll 4
    for (int e = 0; e < 16; ++e) {
        int p = e * 256 + t;
        calc_idx(ob, p, i1, i2, i3, i4, d1, d2, d3, d4);
        float v4 = vc[i4];                      // v4 = gather(v3, cx, fy)
        float res = d3 * (d1 * va[p] + d2 * vb[p]) + d4 * (d1 * v4 + d2 * vc[p]);
        dst[((size_t)b * 256 + c) * CHW + p] = fmaxf(res, 0.0f);
    }
}

// ---------------------------------------------------------------------------
extern "C" void kernel_launch(void* const* d_in, const int* in_sizes, int n_in,
                              void* d_out, int out_size, void* d_ws, size_t ws_size,
                              hipStream_t stream) {
    const float* x  = (const float*)d_in[0];
    const float* w1 = (const float*)d_in[1];
    const float* g1 = (const float*)d_in[2];
    const float* b1 = (const float*)d_in[3];
    const float* m1 = (const float*)d_in[4];
    const float* v1 = (const float*)d_in[5];
    const float* w2 = (const float*)d_in[6];
    const float* g2 = (const float*)d_in[7];
    const float* b2 = (const float*)d_in[8];
    const float* m2 = (const float*)d_in[9];
    const float* v2 = (const float*)d_in[10];
    float* out = (float*)d_out;

    // workspace layout (bytes), all 256-aligned:
    //   xb  bf16 [256][32768]              16,777,216
    //   w1b bf16 [256][2304]                1,179,648
    //   w2b bf16 [512][2304]                2,359,296
    //   o1  f32  [256][32768]              33,554,432
    //   o2  f32  [512][32768]              67,108,864   (total ~121 MB)
    char* ws = (char*)d_ws;
    bf16*  xb  = (bf16*)(ws);
    bf16*  w1b = (bf16*)(ws + 16777216);
    bf16*  w2b = (bf16*)(ws + 16777216 + 1179648);
    float* o1  = (float*)(ws + 20316160);
    float* o2  = (float*)(ws + 53870592);

    k_convert_x<<<32768, 256, 0, stream>>>(x, xb);
    k_convert_w<<<2304, 256, 0, stream>>>(w1, w1b);   // CO=256
    k_convert_w<<<4608, 256, 0, stream>>>(w2, w2b);   // CO=512
    k_conv_bn<<<dim3(NTOT / TILE_N, 2), 256, 0, stream>>>(xb, w1b, g1, b1, m1, v1, o1);
    k_conv_bn<<<dim3(NTOT / TILE_N, 4), 256, 0, stream>>>(xb, w2b, g2, b2, m2, v2, o2);
    k_deform<<<2048, 256, 0, stream>>>(o1, o2, out);
}
